// SelfAttention_2869038153790
// MI455X (gfx1250) — matrix operand order repaired and checked by reading, not compiled, not run on hardware
//
#include <hip/hip_runtime.h>

// ---------------------------------------------------------------------------
// CDNA5 (gfx1250) self-attention: bf16 WMMA everywhere, flash-attention core
// with async global->LDS staging of K/V blocks (ASYNCcnt-tracked).
// ---------------------------------------------------------------------------

typedef __attribute__((ext_vector_type(16))) __bf16        v16bf;
typedef __attribute__((ext_vector_type(8)))  float         v8f;
typedef __attribute__((ext_vector_type(4)))  unsigned int  u32x4;
typedef __attribute__((ext_vector_type(8)))  unsigned int  u32x8;
typedef __attribute__((ext_vector_type(4)))  int           i32x4;

union BFrag { u32x8 u; v16bf v; };                          // 32B contiguous B tile
union AFrag { struct { u32x4 lo; u32x4 hi; } u; v16bf v; }; // 2x16B A tile

#define WMMA_BF16(a, b, c) \
  __builtin_amdgcn_wmma_f32_16x16x32_bf16(false, (a), false, (b), (short)0, (c), false, false)

// gfx1250 async global->LDS copy. Signature (from compiler diagnostic):
//   (i32x4 AS(1)*, i32x4 AS(3)*, imm offset, imm cpol)
#if __has_builtin(__builtin_amdgcn_global_load_async_to_lds_b128)
#define HAVE_ASYNC_LDS 1
typedef __attribute__((address_space(1))) i32x4* gv4i_ptr;
typedef __attribute__((address_space(3))) i32x4* lv4i_ptr;
#define CP16_TO_LDS(dst, src) \
  __builtin_amdgcn_global_load_async_to_lds_b128((gv4i_ptr)(src), (lv4i_ptr)(dst), 0, 0)
#else
#define HAVE_ASYNC_LDS 0
#define CP16_TO_LDS(dst, src) \
  (*reinterpret_cast<u32x4*>(dst) = *reinterpret_cast<const u32x4*>(src))
#endif

// Problem constants
#define BATCH 8
#define SEQ   1024
#define KDIM  256
#define HEADS 8
#define BT    (BATCH * SEQ)      // 8192 rows of x
#define HK    (HEADS * KDIM)     // 2048 projection cols

// ---------------------------------------------------------------------------
// Converts
// ---------------------------------------------------------------------------
__global__ void cvt_f32_bf16(const float* __restrict__ in, __bf16* __restrict__ out, int n) {
  int i = blockIdx.x * blockDim.x + threadIdx.x;
  if (i < n) out[i] = (__bf16)in[i];
}

// in: [rows][cols] f32 -> out: [cols][rows] bf16 (W stored (in,out); we want (out,in))
__global__ void cvt_transpose_bf16(const float* __restrict__ in, __bf16* __restrict__ out,
                                   int rows, int cols) {
  int i = blockIdx.x * blockDim.x + threadIdx.x;
  if (i < rows * cols) {
    int r = i / cols, c = i % cols;
    out[(size_t)c * rows + r] = (__bf16)in[i];
  }
}

// ---------------------------------------------------------------------------
// Projection GEMM: X[8192][256] @ W -> 8192x2048, per-wave 32x64 tile
// (2 A-frags reused across 4 B-frags -> 8 WMMAs per 6 fragment loads).
// mode 0: store [b][h][t][dim] * scale (Q, K with k^-1/4)
// mode 1: store [b][h][dim][t]         (V transposed for att@V B-fragments)
// ---------------------------------------------------------------------------
__global__ void gemm_proj(const __bf16* __restrict__ X,   // [8192][256]
                          const __bf16* __restrict__ Wt,  // [2048][256]
                          __bf16* __restrict__ Out,
                          int mode, float scale) {
  const int lane = threadIdx.x & 31;
  const int wave = threadIdx.x >> 5;
  const int n0   = blockIdx.x * 64;
  const int m0   = blockIdx.y * 128 + wave * 32;
  const int mr   = lane & 15;
  const int half = lane >> 4;

  v8f acc[2][4] = {};
  for (int kb = 0; kb < KDIM; kb += 32) {
    v16bf a[2], b[4];
#pragma unroll
    for (int i = 0; i < 2; ++i) {
      const __bf16* p = X + (size_t)(m0 + i * 16 + mr) * KDIM + kb;
      AFrag f;
      f.u.lo = *reinterpret_cast<const u32x4*>(p + 16 * half);
      f.u.hi = *reinterpret_cast<const u32x4*>(p + 16 * half + 8);
      a[i] = f.v;
    }
#pragma unroll
    for (int j = 0; j < 4; ++j) {
      const __bf16* q = Wt + (size_t)(n0 + j * 16 + mr) * KDIM + kb + 16 * half;
      BFrag g;
      g.u = *reinterpret_cast<const u32x8*>(q);
      b[j] = g.v;
    }
#pragma unroll
    for (int i = 0; i < 2; ++i)
#pragma unroll
      for (int j = 0; j < 4; ++j)
        acc[i][j] = WMMA_BF16(a[i], b[j], acc[i][j]);
  }

#pragma unroll
  for (int i = 0; i < 2; ++i) {
#pragma unroll
    for (int j = 0; j < 4; ++j) {
#pragma unroll
      for (int r = 0; r < 8; ++r) {
        int row = m0 + i * 16 + r + 8 * half;   // = b*1024 + t
        int col = n0 + j * 16 + mr;             // = h*256 + dim
        int bb = row >> 10, tt = row & (SEQ - 1);
        int hh = col >> 8,  dd = col & (KDIM - 1);
        float v = acc[i][j][r] * scale;
        size_t idx;
        if (mode == 0)
          idx = (((size_t)(bb * HEADS + hh) * SEQ) + tt) * KDIM + dd;
        else
          idx = (((size_t)(bb * HEADS + hh) * KDIM) + dd) * SEQ + tt;
        Out[idx] = (__bf16)v;
      }
    }
  }
}

// ---------------------------------------------------------------------------
// Flash attention: block = 4 waves sharing one (b,h); each wave = 16 query
// rows x full head dim. 32 keys/iter: K/V block staged into LDS once per
// block via async global->LDS copies, then 16 score WMMAs + online softmax +
// 16 PV WMMAs, all operand fragments read from padded LDS.
// ---------------------------------------------------------------------------
__global__ void flash_attn(const __bf16* __restrict__ Q,   // [64][1024][256] (pre-scaled)
                           const __bf16* __restrict__ K,   // [64][1024][256] (pre-scaled)
                           const __bf16* __restrict__ VT,  // [64][256][1024]
                           __bf16* __restrict__ O) {       // [8][1024][2048]
  __shared__ __align__(16) __bf16 Kl[32][264];   // 32 keys x 256 dims (+8 pad)
  __shared__ __align__(16) __bf16 Vl[256][40];   // 256 dims x 32 keys (+8 pad)
  __shared__ __align__(16) __bf16 Plds[4][16][40];
  const int tid  = threadIdx.x;
  const int lane = tid & 31;
  const int wave = tid >> 5;
  const int bh   = blockIdx.x;                     // 0..63 = b*8 + h
  const int q0   = (blockIdx.y * 4 + wave) * 16;
  const int mr   = lane & 15;
  const int half = lane >> 4;

  const __bf16* Qh = Q  + (size_t)bh * SEQ * KDIM;
  const __bf16* Kh = K  + (size_t)bh * SEQ * KDIM;
  const __bf16* Vh = VT + (size_t)bh * KDIM * SEQ;

  // Preload Q fragments once (reused for all 32 key blocks)
  v16bf qf[8];
#pragma unroll
  for (int c = 0; c < 8; ++c) {
    const __bf16* p = Qh + (size_t)(q0 + mr) * KDIM + c * 32;
    AFrag f;
    f.u.lo = *reinterpret_cast<const u32x4*>(p + 16 * half);
    f.u.hi = *reinterpret_cast<const u32x4*>(p + 16 * half + 8);
    qf[c] = f.v;
  }

  v8f acc[16] = {};
  float mrun[8], lrun[8];
#pragma unroll
  for (int r = 0; r < 8; ++r) { mrun[r] = -1e30f; lrun[r] = 0.0f; }

  for (int s0 = 0; s0 < SEQ; s0 += 32) {
    // ---- stage K block (contiguous 16KB) and V block into LDS, 16B/lane/op
#pragma unroll
    for (int c = 0; c < 8; ++c) {
      int chunk = tid + c * 128;         // 0..1023: 32 rows x 32 chunks
      int row = chunk >> 5, sub = chunk & 31;
      CP16_TO_LDS(&Kl[row][sub * 8], Kh + (size_t)(s0 + row) * KDIM + sub * 8);
    }
#pragma unroll
    for (int c = 0; c < 8; ++c) {
      int chunk = tid + c * 128;         // 0..1023: 256 rows x 4 chunks
      int row = chunk >> 2, sub = chunk & 3;
      CP16_TO_LDS(&Vl[row][sub * 8], Vh + (size_t)row * SEQ + s0 + sub * 8);
    }
#if HAVE_ASYNC_LDS
    asm volatile("s_wait_asynccnt 0x0" ::: "memory");
#endif
    __syncthreads();

    // prefetch next K block into caches while we compute on this one
    if (s0 + 32 < SEQ)
      __builtin_prefetch(Kh + (size_t)(s0 + 32) * KDIM + lane * 32, 0, 1);

    // ---- scores: Q(16x256) @ K^T(256x32) -> two 16x16 tiles (from LDS)
    v8f sc[2] = {};
#pragma unroll
    for (int c = 0; c < 8; ++c) {
#pragma unroll
      for (int j = 0; j < 2; ++j) {
        BFrag g;
        g.u = *reinterpret_cast<const u32x8*>(&Kl[j * 16 + mr][c * 32 + 16 * half]);
        sc[j] = WMMA_BF16(qf[c], g.v, sc[j]);
      }
    }

    // ---- online softmax (rows r+8*half live across the 16 lanes of each half)
#pragma unroll
    for (int r = 0; r < 8; ++r) {
      float mx = fmaxf(sc[0][r], sc[1][r]);
#pragma unroll
      for (int d = 8; d >= 1; d >>= 1) mx = fmaxf(mx, __shfl_xor(mx, d, 16));
      float mnew    = fmaxf(mrun[r], mx);
      float rescale = __expf(mrun[r] - mnew);
      float p0 = __expf(sc[0][r] - mnew);
      float p1 = __expf(sc[1][r] - mnew);
      float rs = p0 + p1;
#pragma unroll
      for (int d = 8; d >= 1; d >>= 1) rs += __shfl_xor(rs, d, 16);
      lrun[r] = lrun[r] * rescale + rs;
      mrun[r] = mnew;
#pragma unroll
      for (int n = 0; n < 16; ++n) acc[n][r] *= rescale;
      // P tile (D layout: row r+8*half, col mr / 16+mr) -> LDS row-major 16x32
      Plds[wave][r + 8 * half][mr]      = (__bf16)p0;
      Plds[wave][r + 8 * half][16 + mr] = (__bf16)p1;
    }

    // ---- re-read P as an A fragment (16x32); same-wave LDS RAW, DScnt-ordered
    AFrag pf;
    pf.u.lo = *reinterpret_cast<const u32x4*>(&Plds[wave][mr][16 * half]);
    pf.u.hi = *reinterpret_cast<const u32x4*>(&Plds[wave][mr][16 * half + 8]);

    // ---- att @ V over all 16 head-dim tiles (V rows contiguous in key dim)
#pragma unroll
    for (int n = 0; n < 16; ++n) {
      BFrag g;
      g.u = *reinterpret_cast<const u32x8*>(&Vl[n * 16 + mr][16 * half]);
      acc[n] = WMMA_BF16(pf.v, g.v, acc[n]);
    }
    __syncthreads();   // protect Kl/Vl before next iteration's staging
  }

  // ---- epilogue: divide by l, store bf16 into [b][t][h*256]
  const int b = bh >> 3, h = bh & 7;
#pragma unroll
  for (int r = 0; r < 8; ++r) {
    float invl = 1.0f / lrun[r];
    int tq = q0 + r + 8 * half;
    __bf16* orow = O + ((size_t)(b * SEQ + tq)) * HK + h * KDIM;
#pragma unroll
    for (int n = 0; n < 16; ++n)
      orow[n * 16 + mr] = (__bf16)(acc[n][r] * invl);
  }
}

// ---------------------------------------------------------------------------
// Unify GEMM: AttO[8192][2048] @ Wu(2048x256) + bias -> f32 out [8192][256]
// (per-wave 32x64 tile)
// ---------------------------------------------------------------------------
__global__ void gemm_unify(const __bf16* __restrict__ A,   // [8192][2048]
                           const __bf16* __restrict__ Wt,  // [256][2048]
                           const float* __restrict__ bias, // [256]
                           float* __restrict__ Out) {      // [8192][256]
  const int lane = threadIdx.x & 31;
  const int wave = threadIdx.x >> 5;
  const int n0   = blockIdx.x * 64;
  const int m0   = blockIdx.y * 128 + wave * 32;
  const int mr   = lane & 15;
  const int half = lane >> 4;

  v8f acc[2][4] = {};
  for (int kb = 0; kb < HK; kb += 32) {
    v16bf a[2], b[4];
#pragma unroll
    for (int i = 0; i < 2; ++i) {
      const __bf16* p = A + (size_t)(m0 + i * 16 + mr) * HK + kb;
      AFrag f;
      f.u.lo = *reinterpret_cast<const u32x4*>(p + 16 * half);
      f.u.hi = *reinterpret_cast<const u32x4*>(p + 16 * half + 8);
      a[i] = f.v;
    }
#pragma unroll
    for (int j = 0; j < 4; ++j) {
      const __bf16* q = Wt + (size_t)(n0 + j * 16 + mr) * HK + kb + 16 * half;
      BFrag g;
      g.u = *reinterpret_cast<const u32x8*>(q);
      b[j] = g.v;
    }
#pragma unroll
    for (int i = 0; i < 2; ++i)
#pragma unroll
      for (int j = 0; j < 4; ++j)
        acc[i][j] = WMMA_BF16(a[i], b[j], acc[i][j]);
  }

#pragma unroll
  for (int i = 0; i < 2; ++i) {
#pragma unroll
    for (int j = 0; j < 4; ++j) {
#pragma unroll
      for (int r = 0; r < 8; ++r) {
        int row = m0 + i * 16 + r + 8 * half;
        int col = n0 + j * 16 + mr;
        Out[(size_t)row * KDIM + col] = acc[i][j][r] + bias[col];
      }
    }
  }
}

// ---------------------------------------------------------------------------
// Host launcher
// ---------------------------------------------------------------------------
extern "C" void kernel_launch(void* const* d_in, const int* in_sizes, int n_in,
                              void* d_out, int out_size, void* d_ws, size_t ws_size,
                              hipStream_t stream) {
  const float* x  = (const float*)d_in[0];
  const float* Wq = (const float*)d_in[1];
  const float* Wk = (const float*)d_in[2];
  const float* Wv = (const float*)d_in[3];
  const float* Wu = (const float*)d_in[4];
  const float* bu = (const float*)d_in[5];
  float* out = (float*)d_out;

  char* ws = (char*)d_ws;
  __bf16* Xb   = (__bf16*)ws; ws += (size_t)BT * KDIM * 2;          // 4 MiB
  __bf16* Wqt  = (__bf16*)ws; ws += (size_t)HK * KDIM * 2;          // 1 MiB
  __bf16* Wkt  = (__bf16*)ws; ws += (size_t)HK * KDIM * 2;
  __bf16* Wvt  = (__bf16*)ws; ws += (size_t)HK * KDIM * 2;
  __bf16* Wut  = (__bf16*)ws; ws += (size_t)HK * KDIM * 2;
  __bf16* Qb   = (__bf16*)ws; ws += (size_t)BATCH * HEADS * SEQ * KDIM * 2;  // 32 MiB
  __bf16* Kb   = (__bf16*)ws; ws += (size_t)BATCH * HEADS * SEQ * KDIM * 2;
  __bf16* VTb  = (__bf16*)ws; ws += (size_t)BATCH * HEADS * KDIM * SEQ * 2;
  __bf16* AttO = (__bf16*)ws; ws += (size_t)BT * HK * 2;                    // 32 MiB

  const float inv4 = 0.25f;  // 256^(-1/4)

  // Precision conversion + weight transposes
  cvt_f32_bf16<<<(BT * KDIM) / 256, 256, 0, stream>>>(x, Xb, BT * KDIM);
  cvt_transpose_bf16<<<(KDIM * HK) / 256, 256, 0, stream>>>(Wq, Wqt, KDIM, HK);
  cvt_transpose_bf16<<<(KDIM * HK) / 256, 256, 0, stream>>>(Wk, Wkt, KDIM, HK);
  cvt_transpose_bf16<<<(KDIM * HK) / 256, 256, 0, stream>>>(Wv, Wvt, KDIM, HK);
  cvt_transpose_bf16<<<(HK * KDIM) / 256, 256, 0, stream>>>(Wu, Wut, HK, KDIM);

  dim3 blk(128);  // 4 waves
  // Projections: Q,K scaled by k^-1/4; V stored transposed per head
  gemm_proj<<<dim3(HK / 64, BT / 128), blk, 0, stream>>>(Xb, Wqt, Qb, 0, inv4);
  gemm_proj<<<dim3(HK / 64, BT / 128), blk, 0, stream>>>(Xb, Wkt, Kb, 0, inv4);
  gemm_proj<<<dim3(HK / 64, BT / 128), blk, 0, stream>>>(Xb, Wvt, VTb, 1, 1.0f);

  // Streaming softmax(QK^T)V — score matrix never hits memory
  flash_attn<<<dim3(BATCH * HEADS, SEQ / 64), blk, 0, stream>>>(Qb, Kb, VTb, AttO);

  // Unify heads + bias -> f32 output
  gemm_unify<<<dim3(KDIM / 64, BT / 128), blk, 0, stream>>>(AttO, Wut, bu, out);
}
